// BOTNet_37434934952454
// MI455X (gfx1250) — matrix operand order repaired and checked by reading, not compiled, not run on hardware
//
#include <hip/hip_runtime.h>
#include <hip/hip_bf16.h>

// ---------------------------------------------------------------------------
// BOTNet-lite on MI455X (gfx1250, wave32).
// Only the m=0 channel of feats is consumed downstream and Y[:,0]==1, so
// spherical harmonics drop out of energy AND forces entirely; forces flow
// only through the radial features.  Node-level GEMMs use
// v_wmma_f32_16x16x32_f16 with fp32 accumulate.  A-fragments are loaded
// directly from global (2 contiguous float4 runs per fragment); B-fragments
// are staged once in LDS in fragment-major order (one aligned 32B v16h read
// per fragment).  Edge message passing: LDS-staged weights + f32 atomics.
// ---------------------------------------------------------------------------

typedef __attribute__((ext_vector_type(16))) _Float16 v16h;
typedef __attribute__((ext_vector_type(8)))  float    v8f;

#define PI_F      3.14159265358979323846f
#define INV_RMAX  0.2f
#define C_BESS    0.632455532033675866f   // sqrt(2/5)

__device__ __forceinline__ float silu_f(float x){
  float s = 1.f/(1.f+__expf(-x)); return x*s;
}
__device__ __forceinline__ float silu_grad_f(float x){
  float s = 1.f/(1.f+__expf(-x)); return s*(1.f + x*(1.f-s));
}

// Bessel * polynomial envelope (p=5)
__device__ __forceinline__ void radial_feat(float r, float* Rf){
  float u  = r*INV_RMAX;
  float rm = fmaxf(r, 1e-9f);
  float inv = 1.f/rm;
  float env = 0.f;
  if (u < 1.f){
    float u2=u*u, u4=u2*u2, u5=u4*u, u6=u5*u, u7=u6*u;
    env = 1.f - 21.f*u5 + 35.f*u6 - 15.f*u7;
  }
  #pragma unroll
  for (int j=0;j<8;++j){
    float a = (float)(j+1)*PI_F*INV_RMAX;
    Rf[j] = C_BESS * __sinf(a*r) * inv * env;
  }
}

// sum_j dRf[j] * d(Rfeat_j)/dr
__device__ __forceinline__ float radial_dr_dot(float r, const float* dRf){
  float u  = r*INV_RMAX;
  float rm = fmaxf(r, 1e-9f);
  float inv = 1.f/rm;
  float step = (r >= 1e-9f) ? 1.f : 0.f;
  float env = 0.f, denv_dr = 0.f;
  if (u < 1.f){
    float u2=u*u, u4=u2*u2, u5=u4*u, u6=u5*u, u7=u6*u;
    env     = 1.f - 21.f*u5 + 35.f*u6 - 15.f*u7;
    denv_dr = (-105.f*u4 + 210.f*u5 - 105.f*u6) * INV_RMAX;
  }
  float acc = 0.f;
  #pragma unroll
  for (int j=0;j<8;++j){
    float a = (float)(j+1)*PI_F*INV_RMAX;
    float s, c; __sincosf(a*r, &s, &c);
    float d = C_BESS * ( a*c*inv*env - s*inv*inv*step*env + s*inv*denv_dr );
    acc += dRf[j]*d;
  }
  return acc;
}

// -------------------- WMMA fragment helpers (wave32) -----------------------
// A (16xK f16): lane L holds row m=L&15; element i -> K = i + 8*((i>>3)+(L>>4)).
//   For fixed (kc, hb=L>>4) that is two contiguous 8-float runs at
//   K0 = kc*32 + 8*hb and K0+16; every element is owned by exactly one lane,
//   so fragments load straight from global as float4 pairs.
// B (Kx16 f16): lane L holds column n=L&15; element i -> K = kc*32+i+8*((i>>3)+(L>>4)), same as A.
//   Staged once in LDS in fragment-major order: one aligned v16h per read.
// C (16x16 f32): element v, lane L -> row v+8*(L>>4), col L&15.

__device__ __forceinline__ void put8(v16h& a, int base, float4 x, float4 y, float sc){
  a[base+0]=(_Float16)(x.x*sc); a[base+1]=(_Float16)(x.y*sc);
  a[base+2]=(_Float16)(x.z*sc); a[base+3]=(_Float16)(x.w*sc);
  a[base+4]=(_Float16)(y.x*sc); a[base+5]=(_Float16)(y.y*sc);
  a[base+6]=(_Float16)(y.z*sc); a[base+7]=(_Float16)(y.w*sc);
}

// A row assembled from [M/16 | NS] (128-K fused forward GEMM)
__device__ __forceinline__ v16h make_a_fwd(const float* __restrict__ M,
                                           const float* __restrict__ NSp,
                                           int n, int kc, int hb, bool valid){
  v16h a;
  if (!valid){
    #pragma unroll
    for (int i=0;i<16;++i) a[i]=(_Float16)0.f;
    return a;
  }
  int K0 = kc*32 + 8*hb;
  #pragma unroll
  for (int h=0; h<2; ++h){
    int K = K0 + 16*h;                     // 8-float run, never straddles 64
    const float* src; float sc;
    if (K < 64){ src = M   + (size_t)n*64 + K;      sc = 1.f/16.f; }
    else       { src = NSp + (size_t)n*64 + (K-64); sc = 1.f;      }
    float4 x = ((const float4*)src)[0];
    float4 y = ((const float4*)src)[1];
    put8(a, 8*h, x, y, sc);
  }
  return a;
}

// A row from a single N x 64 source (backward GEMMs, K=64)
__device__ __forceinline__ v16h make_a_plain(const float* __restrict__ Gp,
                                             int n, int kc, int hb, bool valid){
  v16h a;
  if (!valid){
    #pragma unroll
    for (int i=0;i<16;++i) a[i]=(_Float16)0.f;
    return a;
  }
  int K0 = kc*32 + 8*hb;
  #pragma unroll
  for (int h=0; h<2; ++h){
    const float* src = Gp + (size_t)n*64 + K0 + 16*h;
    float4 x = ((const float4*)src)[0];
    float4 y = ((const float4*)src)[1];
    put8(a, 8*h, x, y, 1.f);
  }
  return a;
}

// -------------------- embedding + e0 ---------------------------------------
__global__ __launch_bounds__(256) void embed_kernel(
    const float* __restrict__ attrs, const float* __restrict__ Wemb,
    const float* __restrict__ ae, const int* __restrict__ batch,
    float* __restrict__ ns, float* __restrict__ contribs, int N)
{
  int id = blockIdx.x*256 + threadIdx.x;
  if (id >= N*64) return;
  int n = id >> 6, k = id & 63;
  float acc = 0.f;
  #pragma unroll
  for (int e=0;e<10;++e) acc += attrs[n*10+e]*Wemb[e*64+k];
  ns[id] = acc;
  if (k == 0){
    float e0 = 0.f;
    #pragma unroll
    for (int e=0;e<10;++e) e0 += attrs[n*10+e]*ae[e];
    atomicAdd(&contribs[batch[n]*3+0], e0);
  }
}

// -------------------- per-edge forward: scatter ws into M ------------------
__global__ __launch_bounds__(256) void edge_fwd_kernel(
    const float* __restrict__ pos, const float* __restrict__ shifts,
    const int* __restrict__ eidx, int E,
    const float* __restrict__ ns,           // sender features N x 64
    const float* __restrict__ Wr1,          // 8 x 32
    const float* __restrict__ Wr2,          // 32 x 64
    float* __restrict__ Magg,               // N x 64 (atomic)
    float* __restrict__ r_out, int store_r)
{
  __shared__ float sWr1[8*32];
  __shared__ float sWr2[32*64];
  for (int i=threadIdx.x;i<8*32;  i+=256) sWr1[i]=Wr1[i];
  for (int i=threadIdx.x;i<32*64; i+=256) sWr2[i]=Wr2[i];
  __syncthreads();
  int e = blockIdx.x*256 + threadIdx.x;
  if (e >= E) return;
  int s = eidx[e], t = eidx[E+e];
  float vx = pos[t*3+0]-pos[s*3+0]+shifts[e*3+0];
  float vy = pos[t*3+1]-pos[s*3+1]+shifts[e*3+1];
  float vz = pos[t*3+2]-pos[s*3+2]+shifts[e*3+2];
  float r = sqrtf(vx*vx+vy*vy+vz*vz + 1e-18f);
  if (store_r) r_out[e] = r;
  float Rf[8]; radial_feat(r, Rf);
  float H[32];
  #pragma unroll
  for (int j=0;j<32;++j){
    float z=0.f;
    #pragma unroll
    for (int i=0;i<8;++i) z += Rf[i]*sWr1[i*32+j];
    H[j] = silu_f(z);
  }
  const float* nsrow = ns + (size_t)s*64;
  float* Mrow = Magg + (size_t)t*64;
  #pragma unroll 4
  for (int k=0;k<64;++k){
    float Rk=0.f;
    #pragma unroll
    for (int j=0;j<32;++j) Rk += H[j]*sWr2[j*64+k];
    atomicAdd(&Mrow[k], Rk*nsrow[k]);
  }
}

// -------------------- node GEMM fwd: f = (M/16)@Wmix0 + NS@Wself -----------
__global__ __launch_bounds__(256) void gemm_fwd_kernel(
    const float* __restrict__ Msum, const float* __restrict__ NS,
    const float* __restrict__ Wmix0, const float* __restrict__ Wself,
    float* __restrict__ Fout, int N)
{
  __shared__ __align__(32) _Float16 Blds[4*4*512];   // [kc][t][lane][i], 16KB
  int tid = threadIdx.x;
  for (int idx=tid; idx<4*4*512; idx+=256){
    int i = idx & 15, lane = (idx>>4)&31, f = idx>>9;
    int kc = f>>2, t = f&3;
    int K = kc*32 + (i&7) + 8*(lane>>4) + 16*(i>>3);
    int c = t*16 + (lane&15);
    float v = (K<64) ? Wmix0[K*64+c] : Wself[(K-64)*64+c];
    Blds[idx] = (_Float16)v;
  }
  __syncthreads();
  int wave=tid>>5, lane=tid&31;
  int row0 = (blockIdx.x*8+wave)*16;
  int hb = lane>>4;
  int n = row0 + (lane&15);
  bool valid = (n < N);
  const v16h* Bf = (const v16h*)Blds;
  v8f acc[4] = {};
  #pragma unroll
  for (int kc=0;kc<4;++kc){
    v16h a = make_a_fwd(Msum, NS, valid?n:0, kc, hb, valid);
    #pragma unroll
    for (int t=0;t<4;++t){
      v16h b = Bf[(kc*4+t)*32 + lane];
      acc[t] = __builtin_amdgcn_wmma_f32_16x16x32_f16(false, a, false, b,
                                                      (short)0, acc[t], false, false);
    }
  }
  int nn=lane&15;
  #pragma unroll
  for (int t=0;t<4;++t)
    #pragma unroll
    for (int v=0;v<8;++v){
      int rr = row0 + v + 8*hb;
      if (rr < N) Fout[(size_t)rr*64 + t*16 + nn] = acc[t][v];
    }
}

// -------------------- e1 readout -------------------------------------------
__global__ __launch_bounds__(256) void e1_kernel(
    const float* __restrict__ f1, const float* __restrict__ wread,
    const int* __restrict__ batch, float* __restrict__ contribs, int N)
{
  int n = blockIdx.x*256 + threadIdx.x;
  if (n >= N) return;
  float acc=0.f;
  #pragma unroll 8
  for (int k=0;k<64;++k) acc += f1[(size_t)n*64+k]*wread[k];
  atomicAdd(&contribs[batch[n]*3+1], acc);
}

// -------------------- layer-2 MLP readout + g2 = dE/df2 --------------------
__global__ __launch_bounds__(256) void readout2_kernel(
    const float* __restrict__ f2, const int* __restrict__ batch,
    const float* __restrict__ Wmlp1 /*64x16*/, const float* __restrict__ wmlp2 /*16*/,
    float* __restrict__ contribs, float* __restrict__ g2, int N)
{
  int n = blockIdx.x*256 + threadIdx.x;
  if (n >= N) return;
  const float* f = f2 + (size_t)n*64;
  float z[16] = {};
  for (int k=0;k<64;++k){
    float fv = f[k];
    #pragma unroll
    for (int i=0;i<16;++i) z[i] += fv*Wmlp1[k*16+i];
  }
  float e2=0.f, dz[16];
  #pragma unroll
  for (int i=0;i<16;++i){
    e2    += silu_f(z[i])*wmlp2[i];
    dz[i]  = wmlp2[i]*silu_grad_f(z[i]);
  }
  atomicAdd(&contribs[batch[n]*3+2], e2);
  float* g = g2 + (size_t)n*64;           // may alias f2: all reads done above
  for (int k=0;k<64;++k){
    float acc=0.f;
    #pragma unroll
    for (int i=0;i<16;++i) acc += dz[i]*Wmlp1[k*16+i];
    g[k]=acc;
  }
}

// ---- bwd node GEMM (layer2): [dM1 | gf1] = g2 @ [Wmix1_0^T/16 | Wself1^T] --
__global__ __launch_bounds__(256) void gemm_bwd2_kernel(
    const float* __restrict__ G2, const float* __restrict__ Wmix0,
    const float* __restrict__ Wself, const float* __restrict__ wread,
    float* __restrict__ dM, float* __restrict__ gf, int N)
{
  __shared__ __align__(32) _Float16 Blds[2*8*512];   // [kc][t][lane][i], 16KB
  int tid=threadIdx.x;
  for (int idx=tid; idx<2*8*512; idx+=256){
    int i = idx & 15, lane = (idx>>4)&31, f = idx>>9;
    int kc = f>>3, t = f&7;
    int K = kc*32 + (i&7) + 8*(lane>>4) + 16*(i>>3);      // K < 64 (the contraction index h)
    int c = t*16 + (lane&15);              // output column (0..127)
    float v = (c<64) ? Wmix0[c*64+K]*(1.f/16.f) : Wself[(c-64)*64+K];
    Blds[idx]=(_Float16)v;
  }
  __syncthreads();
  int wave=tid>>5, lane=tid&31; int row0=(blockIdx.x*8+wave)*16;
  int hb=lane>>4;
  int n = row0 + (lane&15);
  bool valid = (n < N);
  const v16h* Bf = (const v16h*)Blds;
  v8f acc[8] = {};
  #pragma unroll
  for (int kc=0;kc<2;++kc){
    v16h a = make_a_plain(G2, valid?n:0, kc, hb, valid);
    #pragma unroll
    for (int t=0;t<8;++t){
      v16h b = Bf[(kc*8+t)*32 + lane];
      acc[t] = __builtin_amdgcn_wmma_f32_16x16x32_f16(false, a, false, b,
                                                      (short)0, acc[t], false, false);
    }
  }
  int nn=lane&15;
  #pragma unroll
  for (int t=0;t<8;++t)
    #pragma unroll
    for (int v=0;v<8;++v){
      int rr=row0+v+8*hb; if (rr>=N) continue;
      int c=t*16+nn; float val=acc[t][v];
      if (c<64) dM[(size_t)rr*64+c]=val;
      else      gf[(size_t)rr*64+(c-64)] = val + wread[c-64]; // + e1 readout grad
    }
}

// ---- bwd node GEMM (layer1): dM0 = gf1 @ Wmix0_0^T / 16 -------------------
__global__ __launch_bounds__(256) void gemm_bwd1_kernel(
    const float* __restrict__ GF, const float* __restrict__ Wmix0,
    float* __restrict__ dM, int N)
{
  __shared__ __align__(32) _Float16 Blds[2*4*512];   // 8KB
  int tid=threadIdx.x;
  for (int idx=tid; idx<2*4*512; idx+=256){
    int i = idx & 15, lane = (idx>>4)&31, f = idx>>9;
    int kc = f>>2, t = f&3;
    int K = kc*32 + (i&7) + 8*(lane>>4) + 16*(i>>3);
    int c = t*16 + (lane&15);
    Blds[idx]=(_Float16)(Wmix0[c*64+K]*(1.f/16.f));
  }
  __syncthreads();
  int wave=tid>>5, lane=tid&31; int row0=(blockIdx.x*8+wave)*16;
  int hb=lane>>4;
  int n = row0 + (lane&15);
  bool valid = (n < N);
  const v16h* Bf = (const v16h*)Blds;
  v8f acc[4] = {};
  #pragma unroll
  for (int kc=0;kc<2;++kc){
    v16h a = make_a_plain(GF, valid?n:0, kc, hb, valid);
    #pragma unroll
    for (int t=0;t<4;++t){
      v16h b = Bf[(kc*4+t)*32 + lane];
      acc[t] = __builtin_amdgcn_wmma_f32_16x16x32_f16(false, a, false, b,
                                                      (short)0, acc[t], false, false);
    }
  }
  int nn=lane&15;
  #pragma unroll
  for (int t=0;t<4;++t)
    #pragma unroll
    for (int v=0;v<8;++v){
      int rr=row0+v+8*hb;
      if (rr<N) dM[(size_t)rr*64 + t*16+nn]=acc[t][v];
    }
}

// -------------------- per-edge backward ------------------------------------
__global__ __launch_bounds__(256) void edge_bwd_kernel(
    const float* __restrict__ pos, const float* __restrict__ shifts,
    const int* __restrict__ eidx, int E,
    const float* __restrict__ ns,           // sender features for this layer
    const float* __restrict__ dM,           // N x 64
    const float* __restrict__ Wr1, const float* __restrict__ Wr2,
    float* __restrict__ gf_out,             // nullable: dE/d(sender feats)
    const float* __restrict__ r_in, float* __restrict__ forces)
{
  __shared__ float sWr1[8*32];
  __shared__ float sWr2[32*64];
  for (int i=threadIdx.x;i<8*32;  i+=256) sWr1[i]=Wr1[i];
  for (int i=threadIdx.x;i<32*64; i+=256) sWr2[i]=Wr2[i];
  __syncthreads();
  int e = blockIdx.x*256 + threadIdx.x;
  if (e >= E) return;
  int s = eidx[e], t = eidx[E+e];
  float vx = pos[t*3+0]-pos[s*3+0]+shifts[e*3+0];
  float vy = pos[t*3+1]-pos[s*3+1]+shifts[e*3+1];
  float vz = pos[t*3+2]-pos[s*3+2]+shifts[e*3+2];
  float r = r_in[e];
  float Rf[8]; radial_feat(r, Rf);
  float Z[32], H[32];
  #pragma unroll
  for (int j=0;j<32;++j){
    float z=0.f;
    #pragma unroll
    for (int i=0;i<8;++i) z += Rf[i]*sWr1[i*32+j];
    Z[j]=z; H[j]=silu_f(z);
  }
  const float* dMrow = dM + (size_t)t*64;
  const float* nsrow = ns + (size_t)s*64;
  float dH[32] = {};
  #pragma unroll 4
  for (int k=0;k<64;++k){
    float Rk=0.f;
    #pragma unroll
    for (int j=0;j<32;++j) Rk += H[j]*sWr2[j*64+k];
    float dws = dMrow[k];
    float dRk = dws * nsrow[k];
    if (gf_out) atomicAdd(&gf_out[(size_t)s*64+k], dws*Rk);
    #pragma unroll
    for (int j=0;j<32;++j) dH[j] += dRk * sWr2[j*64+k];
  }
  float dz[32];
  #pragma unroll
  for (int j=0;j<32;++j) dz[j] = dH[j]*silu_grad_f(Z[j]);
  float dRf[8];
  #pragma unroll
  for (int i=0;i<8;++i){
    float acc=0.f;
    #pragma unroll
    for (int j=0;j<32;++j) acc += dz[j]*sWr1[i*32+j];
    dRf[i]=acc;
  }
  float dEdr = radial_dr_dot(r, dRf);
  float invr = 1.f/r;                       // r >= 1e-9 by construction
  float gx = dEdr*vx*invr, gy = dEdr*vy*invr, gz = dEdr*vz*invr; // dE/dvec
  // vec = pos[t]-pos[s]+shift ; forces = -dE/dpos
  atomicAdd(&forces[(size_t)t*3+0], -gx);
  atomicAdd(&forces[(size_t)t*3+1], -gy);
  atomicAdd(&forces[(size_t)t*3+2], -gz);
  atomicAdd(&forces[(size_t)s*3+0],  gx);
  atomicAdd(&forces[(size_t)s*3+1],  gy);
  atomicAdd(&forces[(size_t)s*3+2],  gz);
}

// -------------------- finalize: total = sum of contribs --------------------
__global__ void finalize_kernel(float* __restrict__ out, int G){
  int g = blockIdx.x*blockDim.x + threadIdx.x;
  if (g >= G) return;
  const float* c = out + G;
  out[g] = c[g*3+0] + c[g*3+1] + c[g*3+2];
}

// ---------------------------------------------------------------------------
extern "C" void kernel_launch(void* const* d_in, const int* in_sizes, int n_in,
                              void* d_out, int out_size, void* d_ws, size_t ws_size,
                              hipStream_t stream)
{
  const float* positions = (const float*)d_in[0];
  const float* node_attrs= (const float*)d_in[1];
  const float* shifts    = (const float*)d_in[2];
  const float* W_embed   = (const float*)d_in[3];
  const float* at_en     = (const float*)d_in[4];
  const float* Wr1_0     = (const float*)d_in[5];
  const float* Wr2_0     = (const float*)d_in[6];
  const float* Wmix_0    = (const float*)d_in[7];   // (4,64,64); block 0 used
  const float* Wself_0   = (const float*)d_in[8];
  const float* w_read_0  = (const float*)d_in[9];
  const float* Wr1_1     = (const float*)d_in[10];
  const float* Wr2_1     = (const float*)d_in[11];
  const float* Wmix_1    = (const float*)d_in[12];
  const float* Wself_1   = (const float*)d_in[13];
  const float* Wmlp1_1   = (const float*)d_in[14];
  const float* wmlp2_1   = (const float*)d_in[15];
  const int*   eidx      = (const int*)d_in[16];
  const int*   batch     = (const int*)d_in[17];

  const int N = in_sizes[0]/3;
  const int E = in_sizes[16]/2;
  const int G = (out_size - 3*N)/4;          // out = [G totals][G*3 contribs][N*3 forces]

  float* out      = (float*)d_out;
  float* contribs = out + G;
  float* forces   = out + 4*G;

  // Workspace layout (floats): 6 node buffers (N*64) + r_edge (E) ~= 80 MB.
  float* ws  = (float*)d_ws;
  size_t NF  = (size_t)N*64;
  float* ns  = ws;            // node_scal (live through edge_bwd1)
  float* f1  = ns  + NF;      // layer-1 feats (live through edge_bwd2)
  float* f2  = f1  + NF;      // layer-2 feats, then reused in-place as g2
  float* M0  = f2  + NF;      // layer-1 message sum, then reused as dM0
  float* M1  = M0  + NF;      // layer-2 message sum, then reused as dM1
  float* gf1 = M1  + NF;      // dE/df1
  float* r_e = gf1 + NF;      // per-edge r

  hipMemsetAsync(d_out, 0, (size_t)out_size*sizeof(float), stream);
  hipMemsetAsync(M0, 0, NF*sizeof(float), stream);
  hipMemsetAsync(M1, 0, NF*sizeof(float), stream);

  int nkBlocks = (N*64 + 255)/256;
  int eBlocks  = (E + 255)/256;
  int nBlocks  = (N + 255)/256;
  int gBlocks  = (N + 127)/128;   // 8 waves x 16 rows per block

  embed_kernel<<<nkBlocks,256,0,stream>>>(node_attrs, W_embed, at_en, batch, ns, contribs, N);

  // ---- layer 1 forward
  edge_fwd_kernel<<<eBlocks,256,0,stream>>>(positions, shifts, eidx, E, ns,
                                            Wr1_0, Wr2_0, M0, r_e, 1);
  gemm_fwd_kernel<<<gBlocks,256,0,stream>>>(M0, ns, Wmix_0, Wself_0, f1, N);
  e1_kernel<<<nBlocks,256,0,stream>>>(f1, w_read_0, batch, contribs, N);

  // ---- layer 2 forward
  edge_fwd_kernel<<<eBlocks,256,0,stream>>>(positions, shifts, eidx, E, f1,
                                            Wr1_1, Wr2_1, M1, r_e, 0);
  gemm_fwd_kernel<<<gBlocks,256,0,stream>>>(M1, f1, Wmix_1, Wself_1, f2, N);
  readout2_kernel<<<nBlocks,256,0,stream>>>(f2, batch, Wmlp1_1, wmlp2_1, contribs,
                                            /*g2 in-place=*/f2, N);

  // ---- backward: layer 2 (dM1 into M1 buffer, gf1 = grad wrt f1)
  gemm_bwd2_kernel<<<gBlocks,256,0,stream>>>(f2, Wmix_1, Wself_1, w_read_0,
                                             /*dM1=*/M1, gf1, N);
  edge_bwd_kernel<<<eBlocks,256,0,stream>>>(positions, shifts, eidx, E, f1,
                                            /*dM=*/M1, Wr1_1, Wr2_1,
                                            /*gf_out=*/gf1, r_e, forces);

  // ---- backward: layer 1 (dM0 into M0 buffer; node_scal has no pos-grad)
  gemm_bwd1_kernel<<<gBlocks,256,0,stream>>>(gf1, Wmix_0, /*dM0=*/M0, N);
  edge_bwd_kernel<<<eBlocks,256,0,stream>>>(positions, shifts, eidx, E, ns,
                                            /*dM=*/M0, Wr1_0, Wr2_0,
                                            /*gf_out=*/nullptr, r_e, forces);

  finalize_kernel<<<1,256,0,stream>>>(out, G);
}
